// LongformerAttention_32839319945665
// MI455X (gfx1250) — matrix operand order repaired and checked by reading, not compiled, not run on hardware
//
#include <hip/hip_runtime.h>
#include <hip/hip_bf16.h>

typedef __bf16 bf16;
typedef __attribute__((ext_vector_type(16))) __bf16        v16bf;
typedef __attribute__((ext_vector_type(8)))  float         v8f;
typedef __attribute__((ext_vector_type(4)))  unsigned int  u32x4;
typedef __attribute__((ext_vector_type(4)))  float         f32x4;
typedef __attribute__((ext_vector_type(4)))  __bf16        bf16x4;
typedef __attribute__((ext_vector_type(4)))  int           i32x4;

// pointer-to-addrspace-qualified int4, as expected by the async-LDS builtins
typedef __attribute__((address_space(1))) i32x4* gbl_i32x4_p;
typedef __attribute__((address_space(3))) i32x4* lds_i32x4_p;

namespace {
constexpr int   S_LEN  = 4096;
constexpr int   HID    = 1024;
constexpr int   NHEAD  = 16;
constexpr int   HDIM   = 64;
constexpr int   HALFW  = 128;
constexpr float NEGBIG = -1.0e30f;
constexpr float SCALE  = 0.125f;   // 1/sqrt(64)
}

union FragB16 { v16bf v; u32x4 q[2]; };

__device__ __forceinline__ v8f v8f_zero() {
    v8f z = {0.f, 0.f, 0.f, 0.f, 0.f, 0.f, 0.f, 0.f};
    return z;
}

__device__ __forceinline__ v8f wmma_bf16(v16bf a, v16bf b, v8f c) {
    return __builtin_amdgcn_wmma_f32_16x16x32_bf16(
        false, a, false, b, (short)0, c, false, false);
}

// ---------------------------------------------------------------------------
// 16-lane butterfly reductions via DPP (pure VALU, no LDS round-trips).
// quad_perm(xor1)=0xB1, quad_perm(xor2)=0x4E, row_half_mirror=0x141,
// row_mirror=0x140.  DPP rows are 16 lanes, so this stays within each
// half-wave group, matching the WMMA C-matrix row layout.
// ---------------------------------------------------------------------------
template <int CTRL>
__device__ __forceinline__ float dpp_mov_f32(float v) {
    union { float f; int i; } u;
    u.f = v;
    u.i = __builtin_amdgcn_update_dpp(u.i, u.i, CTRL, 0xF, 0xF, true);
    return u.f;
}
__device__ __forceinline__ float red_max16(float v) {
    v = fmaxf(v, dpp_mov_f32<0xB1>(v));
    v = fmaxf(v, dpp_mov_f32<0x4E>(v));
    v = fmaxf(v, dpp_mov_f32<0x141>(v));
    v = fmaxf(v, dpp_mov_f32<0x140>(v));
    return v;
}
__device__ __forceinline__ float red_sum16(float v) {
    v += dpp_mov_f32<0xB1>(v);
    v += dpp_mov_f32<0x4E>(v);
    v += dpp_mov_f32<0x141>(v);
    v += dpp_mov_f32<0x140>(v);
    return v;
}

// ---------------------------------------------------------------------------
// Async global->LDS tile staging (CDNA5 GLOBAL_LOAD_ASYNC_TO_LDS_B128,
// tracked by ASYNCcnt). Falls back to a synchronous register bounce if the
// builtin is unavailable on this toolchain.
// ---------------------------------------------------------------------------
#if __has_builtin(__builtin_amdgcn_global_load_async_to_lds_b128)
#define GEMM_ASYNC 1
#else
#define GEMM_ASYNC 0
#endif

__device__ __forceinline__ void tile_copy16(const bf16* g, bf16* l) {
#if GEMM_ASYNC
    __builtin_amdgcn_global_load_async_to_lds_b128(
        (gbl_i32x4_p)(g), (lds_i32x4_p)(l), 0, 0);
#else
    *(u32x4*)l = *(const u32x4*)g;
#endif
}

template <int N>
__device__ __forceinline__ void async_wait_le() {
#if GEMM_ASYNC
#if __has_builtin(__builtin_amdgcn_s_wait_asynccnt)
    __builtin_amdgcn_s_wait_asynccnt((unsigned short)N);
#else
    asm volatile("s_wait_asynccnt %0" :: "n"(N) : "memory");
#endif
#endif
}

// ---------------------------------------------------------------------------
// fp32 -> bf16 conversion (vectorized, n multiple of 4)
// ---------------------------------------------------------------------------
__global__ __launch_bounds__(256)
void k_cvt_bf16(const float* __restrict__ in, bf16* __restrict__ out, int n)
{
    int i = (blockIdx.x * 256 + threadIdx.x) * 4;
    if (i < n) {
        f32x4 v = *(const f32x4*)(in + i);
        bf16x4 o = {(__bf16)v[0], (__bf16)v[1], (__bf16)v[2], (__bf16)v[3]};
        *(bf16x4*)(out + i) = o;
    }
}

// ---------------------------------------------------------------------------
// GEMM: Y[m,n] = sum_k A[m,k] * W[n,k], M=4096, N=K=1024, both row-major.
// 128x128 tile per block, 8 waves (wave = 16 rows x 128 cols), K-step 32,
// double-buffered LDS fed by async global->LDS copies.
// MODE 0: fp32 out row-major [M x N]         (final projection -> d_out)
// MODE 1: bf16 out head-split [h][M][64]     (Q, K)
// MODE 2: bf16 out head-split T [h][64][M]   (V^T)
// ---------------------------------------------------------------------------
template <int MODE>
__global__ __launch_bounds__(256)
void k_gemm_bf16(const bf16* __restrict__ A, const bf16* __restrict__ W,
                 void* __restrict__ out)
{
    constexpr int M = S_LEN, N = HID, Kd = HID;
    __shared__ bf16 As[2][128][32];
    __shared__ bf16 Bs[2][128][32];
    const int tid  = threadIdx.x;
    const int lane = tid & 31;
    const int wave = tid >> 5;
    const int m0   = blockIdx.x * 128;
    const int n0   = blockIdx.y * 128;

    v8f acc[8];
    for (int i = 0; i < 8; ++i) acc[i] = v8f_zero();

    // A fragment (16-bit A 16x32): lane<16 -> row=lane, K 0..15;
    // lane>=16 -> row=lane-16, K 16..31
    const int arow = wave * 16 + (lane & 15);
    const int aKlo = (lane < 16) ? 0 : 16;
    // B fragment (16-bit B 32x16): lane<16 -> col=lane, K 0..15;
    // lane>=16 -> col=lane-16, K 16..31
    const int bKb  = (lane < 16) ? 0 : 16;

    // Cooperative tile staging: each thread owns 2 fixed 16B chunks per array
    // (rows r0 and r0+64, halfword column c0).
    const int r0 = tid >> 2;          // 0..63
    const int c0 = (tid & 3) * 8;     // halfword column
    const bf16* Ap = A + (size_t)(m0 + r0) * Kd + c0;
    const bf16* Wp = W + (size_t)(n0 + r0) * Kd + c0;

    auto issue_stage = [&](int buf, int kk) {
        tile_copy16(Ap + kk,                     &As[buf][r0][c0]);
        tile_copy16(Ap + (size_t)64 * Kd + kk,   &As[buf][r0 + 64][c0]);
        tile_copy16(Wp + kk,                     &Bs[buf][r0][c0]);
        tile_copy16(Wp + (size_t)64 * Kd + kk,   &Bs[buf][r0 + 64][c0]);
    };

    issue_stage(0, 0);
    int stage = 0;
    for (int kk = 0; kk < Kd; kk += 32, stage ^= 1) {
        const bool more = (kk + 32 < Kd);
        if (more) {
            issue_stage(stage ^ 1, kk + 32);      // next stage in flight
            async_wait_le<4>();                   // current stage complete
        } else {
            async_wait_le<0>();
        }
        __syncthreads();

        FragB16 fa;
        fa.q[0] = *(const u32x4*)(&As[stage][arow][aKlo]);
        fa.q[1] = *(const u32x4*)(&As[stage][arow][aKlo + 8]);
        FragB16 fb[8];
        #pragma unroll
        for (int nt = 0; nt < 8; ++nt) {
            const int bcol = nt * 16 + (lane & 15);
            fb[nt].q[0] = *(const u32x4*)(&Bs[stage][bcol][bKb]);
            fb[nt].q[1] = *(const u32x4*)(&Bs[stage][bcol][bKb + 8]);
        }
        #pragma unroll
        for (int nt = 0; nt < 8; ++nt)
            acc[nt] = wmma_bf16(fa.v, fb[nt].v, acc[nt]);
        __syncthreads();
    }

    // C layout: VGPR r holds row r (lanes 0-15) / row r+8 (lanes 16-31), col=lane%16
    const int rowbase = m0 + wave * 16 + ((lane >> 4) << 3);
    #pragma unroll
    for (int nt = 0; nt < 8; ++nt) {
        const int col = n0 + nt * 16 + (lane & 15);
        #pragma unroll
        for (int r = 0; r < 8; ++r) {
            const int row = rowbase + r;
            const float val = acc[nt][r];
            if (MODE == 0) {
                ((float*)out)[(size_t)row * N + col] = val;
            } else if (MODE == 1) {
                const int h = col >> 6, d = col & 63;
                ((bf16*)out)[((size_t)h * M + row) * HDIM + d] = (bf16)val;
            } else {
                const int h = col >> 6, d = col & 63;
                ((bf16*)out)[((size_t)h * HDIM + d) * M + row] = (bf16)val;
            }
        }
    }
}

// ---------------------------------------------------------------------------
// Windowed flash attention. One wave per (head, 16-query block).
// Q,K: [NH][S][64] bf16 ; Vt: [NH][64][S] bf16 ; O: [S][1024] bf16.
// Key tiles: [qb-8 .. qb+8] clipped, plus tile 0 (global column) if excluded.
// Tiles processed in pairs -> K=32 for the P*V WMMA.
// ---------------------------------------------------------------------------
__device__ __forceinline__ int tile_at(int i, int lo) {
    return (lo > 0) ? ((i == 0) ? 0 : (lo + i - 1)) : i;
}

__global__ __launch_bounds__(128)
void k_attn_window(const bf16* __restrict__ Q, const bf16* __restrict__ Km,
                   const bf16* __restrict__ Vt, bf16* __restrict__ O)
{
    __shared__ bf16 Pbuf[4][16][32];   // per-wave P staging (C-layout -> A-layout)
    const int lane = threadIdx.x & 31;
    const int wv   = threadIdx.x >> 5;
    const int task = blockIdx.x * 4 + wv;   // 0..4095
    const int h    = task >> 8;
    const int qb   = task & 255;
    const int q0   = qb * 16;

    const bf16* Qh = Q  + (size_t)h * S_LEN * HDIM;
    const bf16* Kh = Km + (size_t)h * S_LEN * HDIM;
    const bf16* Vh = Vt + (size_t)h * HDIM * S_LEN;

    const int lidx   = lane & 15;
    const int rowoff = (lane >> 4) << 3;
    const int aKlo   = (lane < 16) ? 0 : 16;
    const int bKb    = (lane < 16) ? 0 : 16;

    FragB16 qa0, qa1;
    {
        const bf16* qp = Qh + (size_t)(q0 + lidx) * HDIM;
        qa0.q[0] = *(const u32x4*)(qp + aKlo);
        qa0.q[1] = *(const u32x4*)(qp + aKlo + 8);
        qa1.q[0] = *(const u32x4*)(qp + 32 + aKlo);
        qa1.q[1] = *(const u32x4*)(qp + 32 + aKlo + 8);
    }

    v8f accO[4];
    for (int t = 0; t < 4; ++t) accO[t] = v8f_zero();
    float m[8], l[8];
    #pragma unroll
    for (int r = 0; r < 8; ++r) { m[r] = NEGBIG; l[r] = 0.f; }

    int lo = qb - 8; if (lo < 0) lo = 0;
    int hi = qb + 8; if (hi > 255) hi = 255;
    const int ntile = hi - lo + 1 + ((lo > 0) ? 1 : 0);
    const int npair = (ntile + 1) >> 1;

    for (int p = 0; p < npair; ++p) {
        const int  i1  = 2 * p + 1;
        const int  tA  = tile_at(2 * p, lo);
        const bool dup = (i1 >= ntile);
        const int  tB  = dup ? tA : tile_at(i1, lo);
        const int  k0A = tA * 16, k0B = tB * 16;

        v8f sA = v8f_zero(), sB = v8f_zero();
        {
            const bf16* kp = Kh + (size_t)(k0A + lidx) * HDIM;
            FragB16 fb;
            fb.q[0] = *(const u32x4*)(kp + bKb);
            fb.q[1] = *(const u32x4*)(kp + bKb + 8);
            sA = wmma_bf16(qa0.v, fb.v, sA);
            fb.q[0] = *(const u32x4*)(kp + 32 + bKb);
            fb.q[1] = *(const u32x4*)(kp + 32 + bKb + 8);
            sA = wmma_bf16(qa1.v, fb.v, sA);
        }
        {
            const bf16* kp = Kh + (size_t)(k0B + lidx) * HDIM;
            FragB16 fb;
            fb.q[0] = *(const u32x4*)(kp + bKb);
            fb.q[1] = *(const u32x4*)(kp + bKb + 8);
            sB = wmma_bf16(qa0.v, fb.v, sB);
            fb.q[0] = *(const u32x4*)(kp + 32 + bKb);
            fb.q[1] = *(const u32x4*)(kp + 32 + bKb + 8);
            sB = wmma_bf16(qa1.v, fb.v, sB);
        }

        // masked online softmax; stats per (row, half-wave) to match C layout
        const int colA = k0A + lidx;
        const int colB = k0B + lidx;
        #pragma unroll
        for (int r = 0; r < 8; ++r) {
            const int row = q0 + rowoff + r;
            int dA = row - colA; dA = dA < 0 ? -dA : dA;
            int dB = row - colB; dB = dB < 0 ? -dB : dB;
            const bool vA = (dA <= HALFW) || (colA == 0);
            const bool vB = (!dup) && ((dB <= HALFW) || (colB == 0));
            const float a = vA ? sA[r] * SCALE : NEGBIG;
            const float b = vB ? sB[r] * SCALE : NEGBIG;
            const float mx = red_max16(fmaxf(a, b));
            const float mn = fmaxf(m[r], mx);   // finite: pair 0 contains col 0
            const float pa = __expf(a - mn);    // masked -> 0
            const float pb = __expf(b - mn);
            const float rs = red_sum16(pa + pb);
            const float alpha = __expf(m[r] - mn);
            l[r] = l[r] * alpha + rs;
            m[r] = mn;
            accO[0][r] *= alpha; accO[1][r] *= alpha;
            accO[2][r] *= alpha; accO[3][r] *= alpha;
            Pbuf[wv][rowoff + r][lidx]      = (bf16)pa;
            Pbuf[wv][rowoff + r][lidx + 16] = (bf16)pb;
        }

        FragB16 pf;
        pf.q[0] = *(const u32x4*)(&Pbuf[wv][lidx][aKlo]);
        pf.q[1] = *(const u32x4*)(&Pbuf[wv][lidx][aKlo + 8]);
        const int kbase = (lane < 16) ? k0A : k0B;   // B: K 0-15 = tile A keys
        #pragma unroll
        for (int t = 0; t < 4; ++t) {
            const bf16* vp = Vh + (size_t)(t * 16 + lidx) * S_LEN + kbase;
            FragB16 fv;
            fv.q[0] = *(const u32x4*)(vp);
            fv.q[1] = *(const u32x4*)(vp + 8);
            accO[t] = wmma_bf16(pf.v, fv.v, accO[t]);
        }
    }

    #pragma unroll
    for (int t = 0; t < 4; ++t) {
        const int col = h * HDIM + t * 16 + lidx;
        #pragma unroll
        for (int r = 0; r < 8; ++r) {
            const int row = q0 + rowoff + r;
            O[(size_t)row * HID + col] = (bf16)(accO[t][r] / l[r]);
        }
    }
}

// ---------------------------------------------------------------------------
// Global token row 0: attends ALL keys. One block per head; overwrites O row 0.
// ---------------------------------------------------------------------------
__global__ __launch_bounds__(256)
void k_attn_global_row(const bf16* __restrict__ Q, const bf16* __restrict__ Km,
                       const bf16* __restrict__ Vt, bf16* __restrict__ O)
{
    const int h   = blockIdx.x;
    const int tid = threadIdx.x;
    __shared__ float sc[S_LEN];
    __shared__ float red[256];
    __shared__ float outp[4][64];

    const bf16* Qh = Q  + (size_t)h * S_LEN * HDIM;
    const bf16* Kh = Km + (size_t)h * S_LEN * HDIM;
    const bf16* Vh = Vt + (size_t)h * HDIM * S_LEN;

    float q[64];
    for (int d = 0; d < 64; ++d) q[d] = (float)Qh[d];

    float lmax = NEGBIG;
    for (int j = tid; j < S_LEN; j += 256) {
        const bf16* kr = Kh + (size_t)j * HDIM;
        float s = 0.f;
        for (int d = 0; d < 64; ++d) s += q[d] * (float)kr[d];
        s *= SCALE;
        sc[j] = s;
        lmax = fmaxf(lmax, s);
    }
    red[tid] = lmax; __syncthreads();
    for (int st = 128; st > 0; st >>= 1) {
        if (tid < st) red[tid] = fmaxf(red[tid], red[tid + st]);
        __syncthreads();
    }
    const float mx = red[0]; __syncthreads();

    float lsum = 0.f;
    for (int j = tid; j < S_LEN; j += 256) {
        const float pj = __expf(sc[j] - mx);
        sc[j] = pj;
        lsum += pj;
    }
    red[tid] = lsum; __syncthreads();
    for (int st = 128; st > 0; st >>= 1) {
        if (tid < st) red[tid] += red[tid + st];
        __syncthreads();
    }
    const float inv = 1.0f / red[0]; __syncthreads();

    const int d = tid & 63, part = tid >> 6;   // 4 partial sums per d
    const bf16* vr = Vh + (size_t)d * S_LEN;
    float acc = 0.f;
    for (int j = part * 1024; j < (part + 1) * 1024; ++j)
        acc += sc[j] * (float)vr[j];
    outp[part][d] = acc; __syncthreads();
    if (tid < 64) {
        const float v = (outp[0][tid] + outp[1][tid] + outp[2][tid] + outp[3][tid]) * inv;
        O[(size_t)h * HDIM + tid] = (bf16)v;   // row 0
    }
}

// ---------------------------------------------------------------------------
extern "C" void kernel_launch(void* const* d_in, const int* in_sizes, int n_in,
                              void* d_out, int out_size, void* d_ws, size_t ws_size,
                              hipStream_t stream)
{
    (void)in_sizes; (void)n_in; (void)out_size; (void)ws_size;
    const float* x  = (const float*)d_in[0];
    const float* Wq = (const float*)d_in[1];
    const float* Wk = (const float*)d_in[2];
    const float* Wv = (const float*)d_in[3];
    const float* Wo = (const float*)d_in[4];

    const size_t nX = (size_t)S_LEN * HID;   // 4M elems
    const size_t nW = (size_t)HID * HID;     // 1M elems

    bf16* xb  = (bf16*)d_ws;
    bf16* wqb = xb  + nX;
    bf16* wkb = wqb + nW;
    bf16* wvb = wkb + nW;
    bf16* wob = wvb + nW;
    bf16* qh  = wob + nW;    // [NH][S][64]
    bf16* kh  = qh  + nX;    // [NH][S][64]
    bf16* vth = kh  + nX;    // [NH][64][S]
    bf16* oh  = vth + nX;    // [S][HID]

    k_cvt_bf16<<<(int)(nX / 1024), 256, 0, stream>>>(x,  xb,  (int)nX);
    k_cvt_bf16<<<(int)(nW / 1024), 256, 0, stream>>>(Wq, wqb, (int)nW);
    k_cvt_bf16<<<(int)(nW / 1024), 256, 0, stream>>>(Wk, wkb, (int)nW);
    k_cvt_bf16<<<(int)(nW / 1024), 256, 0, stream>>>(Wv, wvb, (int)nW);
    k_cvt_bf16<<<(int)(nW / 1024), 256, 0, stream>>>(Wo, wob, (int)nW);

    dim3 g(S_LEN / 128, HID / 128);
    k_gemm_bf16<1><<<g, 256, 0, stream>>>(xb, wqb, qh);
    k_gemm_bf16<1><<<g, 256, 0, stream>>>(xb, wkb, kh);
    k_gemm_bf16<2><<<g, 256, 0, stream>>>(xb, wvb, vth);

    k_attn_window<<<(NHEAD * 256) / 4, 128, 0, stream>>>(qh, kh, vth, oh);
    k_attn_global_row<<<NHEAD, 256, 0, stream>>>(qh, kh, vth, oh);

    k_gemm_bf16<0><<<g, 256, 0, stream>>>(oh, wob, d_out);
}
